// BraLM_5085241278626
// MI455X (gfx1250) — matrix operand order repaired and checked by reading, not compiled, not run on hardware
//
#include <hip/hip_runtime.h>
#include <hip/hip_bf16.h>
#include <math.h>

#define RANK 8
#define HID  512
#define EMB  256
#define BS   8
#define SEQ  64
#define NN   32

typedef __attribute__((ext_vector_type(16))) __bf16        v16bf;
typedef __attribute__((ext_vector_type(8)))  float         v8f;
typedef __attribute__((ext_vector_type(8)))  unsigned int  v8u;

static __device__ __forceinline__ unsigned int bfbits(float f) {
  unsigned int u = __builtin_bit_cast(unsigned int, f);
  u += 0x7FFFu + ((u >> 16) & 1u);          // round-to-nearest-even
  return u >> 16;
}
static __device__ __forceinline__ float gelu_exact(float x) {
  return 0.5f * x * (1.0f + erff(x * 0.70710678118654752440f));
}
// pe[p, 2k] = sin(p * 10000^(2k/512)), pe[p, 2k+1] = cos(...)
static __device__ __forceinline__ float pe_val(int p, int h) {
  int kk = h >> 1;
  float dv  = powf(10000.0f, (float)(2 * kk) * (1.0f / 512.0f));
  float ang = (float)p * dv;
  return (h & 1) ? cosf(ang) : sinf(ang);
}

// ---------------------------------------------------------------------------
// One-shot: repack wr_w (4096x256 f32) into WMMA-B-fragment-native bf16.
// Fragment index: [(ct*8 + kt)*32 + lane] -> 16 ushorts (one v8u per lane).
//   ct = wr_w_row/16, colL = row%16; kt = c/32, kk = c%32;
//   B layout: half = (kk>>3)&1, j = ((kk&7)>>1) + 4*(kk>>4), elem = 2j + (kk&1), lane = colL+16*half
// ---------------------------------------------------------------------------
__global__ __launch_bounds__(256)
void pack_wr(const float* __restrict__ wr_w, unsigned short* __restrict__ bpack) {
  int flat = blockIdx.x * 256 + threadIdx.x;   // 4096*256 elements
  int cg = flat >> 8;                          // wr_w row (0..4095) = B column
  int c  = flat & 255;                         // K (0..255)
  int ct = cg >> 4, colL = cg & 15;
  int kt = c >> 5,  kk   = c & 31;
  int half = (kk >> 3) & 1;
  int j    = ((kk & 7) >> 1) + 4 * (kk >> 4);
  int elem = 2 * j + (kk & 1);
  int lane = colL + 16 * half;
  bpack[(size_t)(((ct * 8 + kt) * 32) + lane) * 16 + elem] =
      (unsigned short)bfbits(wr_w[flat]);
}

__global__ __launch_bounds__(256)
void bralm_step(int i,
                const int*   __restrict__ nbr,        // (BS,SEQ,NN,2)
                const float* __restrict__ node_emb,   // (NODES,EMB)
                const float* __restrict__ wl_w,       // (RANK*HID,EMB)
                const float* __restrict__ wl_b,       // (RANK*HID)
                const float* __restrict__ wr_b,       // (RANK*HID)
                const float* __restrict__ bias_base,  // (HID)
                const float* __restrict__ node_bias,  // (NODES,HID)
                const float* __restrict__ positions,  // (PE_LEN)
                const unsigned short* __restrict__ bpack, // packed wr_w bf16
                float* __restrict__ cache,            // (SEQ,BS,HID)
                float* __restrict__ energy_part)      // (BS,8,NN)
{
  const int hg  = blockIdx.x;        // h-slice 0..7 (64 h each)
  const int b   = blockIdx.y;        // batch 0..7
  const int h0  = hg * 64;
  const int tid = threadIdx.x;

  __shared__ float          e_lds[HID];
  __shared__ float          w_lds[SEQ];
  __shared__ float          t_lds[RANK][EMB];
  __shared__ float          bl_lds[RANK];
  __shared__ float          embS[NN][EMB];                 // f32 src embeddings
  __shared__ __align__(32) unsigned short aswz[2 * 8 * 32 * 16]; // A fragments
  __shared__ int            src_lds[NN];
  __shared__ int            tgt_lds[NN];
  __shared__ float          proj_lds[NN][RANK];
  __shared__ float          ener_lds[NN][8];

  // ---- ids + history softmax weights ----
  if (tid < NN) {
    int base = ((b * SEQ + i) * NN + tid) * 2;
    src_lds[tid] = nbr[base];
    tgt_lds[tid] = nbr[base + 1];
  }
  if (i > 0 && tid == 0) {
    float mx = -3.0e38f;
    for (int s = 0; s < i; ++s) mx = fmaxf(mx, positions[s]);
    float den = 0.0f;
    for (int s = 0; s < i; ++s) { float e = expf(positions[s] - mx); w_lds[s] = e; den += e; }
    float inv = 1.0f / den;
    for (int s = 0; s < i; ++s) w_lds[s] *= inv;
  }
  __syncthreads();

  // ---- gather embeddings (tgt directly into A-fragment swizzle), compute e ----
  for (int idx = tid; idx < NN * EMB; idx += 256) {
    int n = idx >> 8, c = idx & 255;
    embS[n][c] = node_emb[(long long)src_lds[n] * EMB + c];
    float tv   = node_emb[(long long)tgt_lds[n] * EMB + c];
    // A (16-bit 16x32) swizzle: invert k0a = (j<4)? 2j+8h : 16+2(j-4)+8h
    int mt = n >> 4, mrow = n & 15;
    int kt = c >> 5, kk = c & 31;
    int half, j;
    if (kk < 16) { half = kk >> 3;            j =      (kk - 8 * half) >> 1; }
    else         { int k2 = kk - 16; half = k2 >> 3; j = 4 + ((k2 - 8 * half) >> 1); }
    int elem = 2 * j + (kk & 1);
    int lane = mrow + 16 * half;
    aswz[(((mt * 8 + kt) * 32) + lane) * 16 + elem] = (unsigned short)bfbits(tv);
  }
  for (int h = tid; h < HID; h += 256) {
    float ev;
    if (i == 0) {
      int   s0  = nbr[b * SEQ * NN * 2];               // neighbor_ids[b,0,0,0]
      float nb  = node_bias[(long long)s0 * HID + h];
      float pe0 = (h & 1) ? 1.0f : 0.0f;               // pe[0,*]
      ev = gelu_exact(gelu_exact(nb) + pe0);
    } else {
      float acc = 0.0f;
      for (int s = 0; s < i; ++s)
        acc = fmaf(w_lds[s], cache[((long long)s * BS + b) * HID + h], acc);
      ev = acc;
    }
    e_lds[h] = ev;
  }
  __syncthreads();

  // ---- factored U path: t[r][c] = sum_h e[h]*wl_w[(r*HID+h)*EMB+c] ----
  {
    int r = tid >> 5, cb = tid & 31;
    float acc[8] = {0, 0, 0, 0, 0, 0, 0, 0};
    const float* wbase = wl_w + (long long)r * HID * EMB + cb;
    for (int h = 0; h < HID; ++h) {
      float eh = e_lds[h];
      const float* row = wbase + (long long)h * EMB;
      #pragma unroll
      for (int q = 0; q < 8; ++q) acc[q] = fmaf(eh, row[q * 32], acc[q]);
    }
    #pragma unroll
    for (int q = 0; q < 8; ++q) t_lds[r][cb + q * 32] = acc[q];
    if (tid < RANK) {
      float s = 0.0f;
      for (int h = 0; h < HID; ++h) s = fmaf(e_lds[h], wl_b[tid * HID + h], s);
      bl_lds[tid] = s;
    }
  }
  __syncthreads();

  // ---- proj[n][r] = bl[r] + embS[n,:] . t[r,:] ----
  {
    int n = tid >> 3, r = tid & 7;
    float s = bl_lds[r];
    for (int c = 0; c < EMB; ++c) s = fmaf(embS[n][c], t_lds[r][c], s);
    proj_lds[n][r] = s;
  }
  __syncthreads();

  // ---- V GEMM via WMMA bf16 (fragment-native loads) + fused epilogue ----
  const int wid  = tid >> 5;
  const int lane = tid & 31;
  const int half = lane >> 4;
  const int lm   = lane & 15;
  const int mt   = wid >> 2;                 // neighbor tile 0..1
  const int hs   = wid & 3;                  // h sub-tile 0..3
  const int hcol = h0 + hs * 16 + lm;        // global h column (0..511)

  const v8u* afr = ((const v8u*)aswz) + (mt * 8) * 32 + lane;   // +kt*32 per kt
  float nxtacc[8] = {0, 0, 0, 0, 0, 0, 0, 0};
  for (int r = 0; r < RANK; ++r) {
    int ct = r * 32 + hg * 4 + hs;           // B 16-col tile index (0..255)
    const v8u* bfr = ((const v8u*)bpack) + (size_t)(ct * 8) * 32 + lane;
    __builtin_prefetch((const void*)(bfr + 32 * 256), 0, 1); // next r tile (speculative)
    v8f acc = {0.f, 0.f, 0.f, 0.f, 0.f, 0.f, 0.f, 0.f};
    #pragma unroll
    for (int kt = 0; kt < 8; ++kt) {
      v8u au = afr[kt * 32];                 // 32B LDS fragment (ds_load_b128 x2)
      v8u bu = bfr[kt * 32];                 // 32B global fragment (b128 x2, L2-hot)
      v16bf Afrag = __builtin_bit_cast(v16bf, au);
      v16bf Bfrag = __builtin_bit_cast(v16bf, bu);
      acc = __builtin_amdgcn_wmma_f32_16x16x32_bf16(
          false, Afrag, false, Bfrag, (short)0, acc, false, false);
    }
    float biasr = wr_b[r * HID + hcol];
    #pragma unroll
    for (int j = 0; j < 8; ++j) {
      int m = mt * 16 + j + 8 * half;        // C/D layout: m = j + 8*half (+ tile)
      nxtacc[j] = fmaf(proj_lds[m][r], acc[j] + biasr, nxtacc[j]);
    }
  }

  const float scale   = 0.35355339059327373f;   // 1/sqrt(RANK)
  const float baseadd = bias_base[hcol] + pe_val(i + 1, hcol);
  float out0v = 0.0f;
  #pragma unroll
  for (int j = 0; j < 8; ++j) {
    int   m  = mt * 16 + j + 8 * half;
    float nb = node_bias[(long long)tgt_lds[m] * HID + hcol];
    float x  = nxtacc[j] * scale + baseadd + nb;
    float ov = gelu_exact(x);
    if (j == 0) out0v = ov;
    float sq = ov * ov;                       // ||out||^2 partial over 16 h lanes
    sq += __shfl_xor(sq, 1, 32);
    sq += __shfl_xor(sq, 2, 32);
    sq += __shfl_xor(sq, 4, 32);
    sq += __shfl_xor(sq, 8, 32);
    if (lm == 0) ener_lds[m][wid] = sq;
  }
  if (mt == 0 && half == 0) {                 // m == 0 (neighbor 0) -> cache
    cache[((long long)i * BS + b) * HID + hcol] = out0v;
  }
  __syncthreads();
  if (tid < NN) {
    float s = 0.0f;
    #pragma unroll
    for (int w = 0; w < 8; ++w) s += ener_lds[tid][w];
    energy_part[(b * 8 + hg) * NN + tid] = s;
  }
}

__global__ void bralm_loss(int i, const float* __restrict__ energy_part,
                           float* __restrict__ loss_part) {
  int b = blockIdx.x;
  int n = threadIdx.x;                        // 32 threads
  float s = 0.0f;
  #pragma unroll
  for (int hg = 0; hg < 8; ++hg) s += energy_part[(b * 8 + hg) * NN + n];
  float en = sqrtf(s);
  float mx = en;
  for (int off = 1; off < 32; off <<= 1) mx = fmaxf(mx, __shfl_xor(mx, off, 32));
  float ex = expf(en - mx);
  float sm = ex;
  for (int off = 1; off < 32; off <<= 1) sm += __shfl_xor(sm, off, 32);
  float en0 = __shfl(en, 0, 32);
  if (n == 0) loss_part[i * BS + b] = -(en0 - mx - logf(sm));
}

__global__ void bralm_final(const float* __restrict__ loss_part,
                            float* __restrict__ out) {
  float a = 0.0f;
  for (int k = 0; k < SEQ * BS; ++k) a += loss_part[k];   // fixed order: deterministic
  out[0] = a * (1.0f / (float)(SEQ * BS));
}

extern "C" void kernel_launch(void* const* d_in, const int* in_sizes, int n_in,
                              void* d_out, int out_size, void* d_ws, size_t ws_size,
                              hipStream_t stream) {
  const int*   nbr       = (const int*)  d_in[0];
  const float* node_emb  = (const float*)d_in[1];
  const float* wl_w      = (const float*)d_in[2];
  const float* wl_b      = (const float*)d_in[3];
  const float* wr_w      = (const float*)d_in[4];
  const float* wr_b      = (const float*)d_in[5];
  const float* bias_base = (const float*)d_in[6];
  const float* node_bias = (const float*)d_in[7];
  const float* positions = (const float*)d_in[8];
  float* out = (float*)d_out;

  // ws layout: bpack first (32B-aligned), then cache / energy / loss
  unsigned short* bpack = (unsigned short*)d_ws;               // 4096*256 bf16 (2 MB)
  float* cache       = (float*)((char*)d_ws + (size_t)RANK * HID * EMB * 2);
  float* energy_part = cache + (size_t)SEQ * BS * HID;         // BS*8*NN
  float* loss_part   = energy_part + BS * 8 * NN;              // SEQ*BS

  pack_wr<<<dim3((RANK * HID * EMB) / 256), 256, 0, stream>>>(wr_w, bpack);

  dim3 grid(8, BS);
  for (int i = 0; i < SEQ; ++i) {
    bralm_step<<<grid, 256, 0, stream>>>(i, nbr, node_emb, wl_w, wl_b, wr_b,
                                         bias_base, node_bias, positions,
                                         bpack, cache, energy_part);
    bralm_loss<<<dim3(BS), dim3(32), 0, stream>>>(i, energy_part, loss_part);
  }
  bralm_final<<<1, 1, 0, stream>>>(loss_part, out);
}
